// FIVO_41772851921257
// MI455X (gfx1250) — matrix operand order repaired and checked by reading, not compiled, not run on hardware
//
#include <hip/hip_runtime.h>
#include <hip/hip_bf16.h>

typedef _Float16 f16;
typedef _Float16 v8h  __attribute__((ext_vector_type(8)));
typedef _Float16 v16h __attribute__((ext_vector_type(16)));
typedef float    v8f  __attribute__((ext_vector_type(8)));

#define USE_ASYNC_LDS 1

#define T_N 64
#define B_N 32
#define X_N 128
#define H_N 512
#define Z_N 64
#define P_N 16
#define M_N (B_N*P_N)   // 512 rows

#define NWG 64
#define TPB 256
#define WPG (TPB/32)
#define GTHREADS (NWG*TPB)

// ---------------- weight region (halves) ----------------
enum : int {
  OFF_Wx1 = 0,
  OFF_Wx2       = OFF_Wx1 + H_N*X_N,
  OFF_Wz        = OFF_Wx2 + H_N*H_N,
  OFF_encW1     = OFF_Wz + H_N*Z_N,
  OFF_encW2     = OFF_encW1 + H_N*2*H_N,
  OFF_encMean   = OFF_encW2 + H_N*H_N,
  OFF_encStd    = OFF_encMean + Z_N*H_N,
  OFF_priorW    = OFF_encStd + Z_N*H_N,
  OFF_priorMean = OFF_priorW + H_N*H_N,
  OFF_priorStd  = OFF_priorMean + Z_N*H_N,
  OFF_decW1     = OFF_priorStd + Z_N*H_N,
  OFF_decW2     = OFF_decW1 + H_N*2*H_N,
  OFF_decMean   = OFF_decW2 + H_N*H_N,
  OFF_Wih       = OFF_decMean + X_N*H_N,
  OFF_Whh       = OFF_Wih + 4*H_N*2*H_N,
  W_TOTAL       = OFF_Whh + 4*H_N*H_N,
};

// ---------------- f16 activation region (halves) ----------------
enum : int {
  AH_XTS   = 0,                      // M x X
  AH_A1    = AH_XTS + M_N*X_N,       // M x H
  AH_PHIX  = AH_A1 + M_N*H_N,
  AH_ENCH1 = AH_PHIX + M_N*H_N,
  AH_ENCH  = AH_ENCH1 + M_N*H_N,
  AH_PRIORH= AH_ENCH + M_N*H_N,
  AH_Z     = AH_PRIORH + M_N*H_N,    // M x Z
  AH_PHIZ  = AH_Z + M_N*Z_N,
  AH_DECH1 = AH_PHIZ + M_N*H_N,
  AH_DECH  = AH_DECH1 + M_N*H_N,
  AH_HCUR  = AH_DECH + M_N*H_N,
  AH_HNEW  = AH_HCUR + M_N*H_N,
  AH_TOTAL = AH_HNEW + M_N*H_N,
};

// ---------------- f32 region (floats) ----------------
enum : int {
  AF_ENCMU  = 0,                     // M x Z
  AF_ENCSTD = AF_ENCMU + M_N*Z_N,
  AF_PRIMU  = AF_ENCSTD + M_N*Z_N,
  AF_PRISTD = AF_PRIMU + M_N*Z_N,
  AF_THETA  = AF_PRISTD + M_N*Z_N,   // M x X
  AF_GATES  = AF_THETA + M_N*X_N,    // M x 4H
  AF_CCUR   = AF_GATES + M_N*4*H_N,  // M x H
  AF_CNEW   = AF_CCUR + M_N*H_N,
  AF_KL     = AF_CNEW + M_N*H_N,     // M
  AF_NLL    = AF_KL + M_N,           // M
  AF_LW     = AF_NLL + M_N,          // B*P
  AF_LHP    = AF_LW + M_N,           // B
  AF_KLACC  = AF_LHP + B_N,
  AF_IWAE   = AF_KLACC + B_N,
  AF_TOTAL  = AF_IWAE + B_N,
};

struct KArgs {
  const float* x; const float* mask;
  const float* bx1; const float* bx2; const float* bz;
  const float* encB1; const float* encB2; const float* encMeanB; const float* encStdB;
  const float* priorB; const float* priorMeanB; const float* priorStdB;
  const float* decB1; const float* decB2; const float* decMeanB;
  const f16* W;
  f16* AH; float* AF; int* idx; unsigned* bar;
  float* out;
};

struct Seg { const f16* A; const f16* W; int ldw; int klen; };

#define EPI_F32   0
#define EPI_RELU  1
#define EPI_SPLUS 2
#define EPI_SIG   3

__device__ __forceinline__ float sigf(float x)   { return 1.0f / (1.0f + __expf(-x)); }
__device__ __forceinline__ float splusf(float x) { return (x > 20.f) ? x : log1pf(__expf(x)); }

__device__ __forceinline__ unsigned hash3(unsigned a, unsigned b, unsigned c) {
  unsigned h = a * 0x9E3779B1u + b * 0x85EBCA77u + c * 0xC2B2AE3Du + 0x27D4EB2Fu;
  h ^= h >> 15; h *= 0x2C1B3C6Du; h ^= h >> 12; h *= 0x297A2D39u; h ^= h >> 15;
  return h;
}
__device__ __forceinline__ float u01(unsigned h) {
  return (float)(h >> 8) * (1.0f/16777216.0f) + (0.5f/16777216.0f);
}
__device__ __forceinline__ float nrmf(unsigned h1, unsigned h2) {
  float u1 = u01(h1), u2 = u01(h2);
  return __fsqrt_rn(-2.0f * __logf(u1)) * __cosf(6.28318530718f * u2);
}
__device__ __forceinline__ float lse16(const float* v, float scale) {
  float m = v[0] * scale;
  #pragma unroll
  for (int p = 1; p < P_N; ++p) m = fmaxf(m, v[p] * scale);
  float s = 0.f;
  #pragma unroll
  for (int p = 0; p < P_N; ++p) s += __expf(v[p] * scale - m);
  return m + __logf(s);
}

// Device-wide barrier (persistent kernel, NWG co-resident workgroups).
__device__ void grid_barrier(unsigned* bar) {
  __syncthreads();
  __threadfence();
  if (threadIdx.x == 0) {
    unsigned* cnt = bar;
    unsigned* gen = bar + 1;
    unsigned g = __hip_atomic_load(gen, __ATOMIC_ACQUIRE, __HIP_MEMORY_SCOPE_AGENT);
    unsigned a = __hip_atomic_fetch_add(cnt, 1u, __ATOMIC_ACQ_REL, __HIP_MEMORY_SCOPE_AGENT);
    if (a == NWG - 1) {
      __hip_atomic_store(cnt, 0u, __ATOMIC_RELAXED, __HIP_MEMORY_SCOPE_AGENT);
      __hip_atomic_fetch_add(gen, 1u, __ATOMIC_ACQ_REL, __HIP_MEMORY_SCOPE_AGENT);
    } else {
      while (__hip_atomic_load(gen, __ATOMIC_ACQUIRE, __HIP_MEMORY_SCOPE_AGENT) == g)
        __builtin_amdgcn_s_sleep(1);
    }
  }
  __syncthreads();
}

// Workgroup-cooperative tiled GEMM, D = A0@W0^T (+A1@W1^T)(+A2@W2^T) + bias.
//  - one M-tile (16 rows x klen) per workgroup unit, staged into LDS via
//    async global->LDS copies, shared by all 8 waves (ds_load_b128 fragments)
//  - each wave computes NT=4 adjacent 16x16 N-tiles (A-fragment register reuse)
__device__ void gemm_run(int lane, int wave, int N,
                         Seg s0, Seg s1, Seg s2,
                         const float* bias, int epi,
                         f16* oH, float* oF, int ldo) {
  __shared__ __attribute__((aligned(16))) f16 ldsA[16 * (H_N + 8)];  // padded rows vs 64-bank conflicts
  const int Ntiles  = N / 16;
  const int nchunks = Ntiles / 4;                       // NT = 4
  const int ngroups = (nchunks + WPG - 1) / WPG;
  const int units   = (M_N / 16) * ngroups;

  for (int u = blockIdx.x; u < units; u += NWG) {
    const int mt    = u / ngroups;
    const int grp   = u - mt * ngroups;
    const int chunk = grp * WPG + wave;
    const bool active = chunk < nchunks;
    const int nbase = chunk * 64;

    v8f acc[4];
    #pragma unroll
    for (int c = 0; c < 4; ++c) {
      float b = (active && bias) ? bias[nbase + c * 16 + (lane & 15)] : 0.0f;
      acc[c] = (v8f){ b, b, b, b, b, b, b, b };
    }

    Seg segs[3] = { s0, s1, s2 };
    #pragma unroll
    for (int si = 0; si < 3; ++si) {
      const Seg s = segs[si];
      if (!s.A) continue;
      const int klen = s.klen;
      const int LDSP = klen + 8;
      // ---- stage A tile (16 x klen) into LDS, whole workgroup ----
      const int cpr = klen / 8;            // 16-byte chunks per row
      const int nch = 16 * cpr;
      for (int ci = threadIdx.x; ci < nch; ci += TPB) {
        const int row = ci / cpr, c8 = ci - row * cpr;
#if USE_ASYNC_LDS
        unsigned ldsoff = (unsigned)(uintptr_t)&ldsA[row * LDSP + c8 * 8];
        unsigned goff   = (unsigned)(((size_t)(mt * 16 + row) * klen + c8 * 8) * sizeof(f16));
        asm volatile("global_load_async_to_lds_b128 %0, %1, %2"
                     :: "v"(ldsoff), "v"(goff), "s"(s.A) : "memory");
#else
        *(v8h*)&ldsA[row * LDSP + c8 * 8] =
            *(const v8h*)(s.A + (size_t)(mt * 16 + row) * klen + c8 * 8);
#endif
      }
#if USE_ASYNC_LDS
      asm volatile("s_wait_asynccnt 0x0" ::: "memory");
#endif
      __syncthreads();
      // ---- accumulate ----
      if (active) {
        const int ka = (lane >> 4) * 16;   // A halves: lanes<16 -> K 0-15; lanes>=16 -> K 16-31
        const int kb = (lane >> 4) * 16;   // B halves: lanes<16 -> K 0-15; lanes>=16 -> K 16-31
        const f16* ldsRow = &ldsA[(lane & 15) * LDSP];
        for (int k0 = 0; k0 < klen; k0 += 32) {
          union { v16h v; struct { v8h lo; v8h hi; } h2; } a;
          a.h2.lo = *(const v8h*)(ldsRow + k0 + ka);        // ds_load_b128
          a.h2.hi = *(const v8h*)(ldsRow + k0 + ka + 8);
          #pragma unroll
          for (int c = 0; c < 4; ++c) {
            const int n = nbase + c * 16 + (lane & 15);
            const f16* bp = s.W + (size_t)n * s.ldw + k0 + kb;
            v16h b = *(const v16h*)bp;
            __builtin_prefetch((const void*)(bp + 32), 0, 0); // global_prefetch_b8
            acc[c] = __builtin_amdgcn_wmma_f32_16x16x32_f16(false, a.v, false, b,
                                                            (short)0, acc[c], false, false);
          }
        }
      }
      __syncthreads();
    }

    // ---- epilogue ----
    if (active) {
      const int rbase = mt * 16 + ((lane >> 4) << 3);  // D: vgpr r -> row r (+8 for lanes>=16)
      #pragma unroll
      for (int c = 0; c < 4; ++c) {
        const int ncol = nbase + c * 16 + (lane & 15);
        #pragma unroll
        for (int r = 0; r < 8; ++r) {
          float v = acc[c][r];
          const size_t o = (size_t)(rbase + r) * ldo + ncol;
          if      (epi == EPI_RELU)  oH[o] = (f16)fmaxf(v, 0.0f);
          else if (epi == EPI_F32)   oF[o] = v;
          else if (epi == EPI_SPLUS) oF[o] = splusf(v);
          else                       oF[o] = sigf(v);
        }
      }
    }
  }
}

__global__ __launch_bounds__(TPB) void fivo_main(KArgs K) {
  __shared__ float s_klmean;
  const int lane = threadIdx.x & 31;
  const int wave = threadIdx.x >> 5;
  const int gtid = blockIdx.x * TPB + threadIdx.x;
  f16* AH = K.AH;
  float* AF = K.AF;
  const f16* W = K.W;
  const float LOGP   = 2.7725887f;   // ln 16
  const float THRESH = 2.0794415f;   // ln 8

  // ---- init carries / accumulators ----
  for (int i = gtid; i < M_N * H_N; i += GTHREADS) { AH[AH_HCUR + i] = (f16)0.f; AF[AF_CCUR + i] = 0.f; }
  for (int i = gtid; i < M_N; i += GTHREADS) AF[AF_LW + i] = 0.f;
  for (int i = gtid; i < B_N; i += GTHREADS) { AF[AF_LHP+i]=0.f; AF[AF_KLACC+i]=0.f; AF[AF_IWAE+i]=0.f; }
  grid_barrier(K.bar);

  for (int t = 0; t < T_N; ++t) {
    const float* xt = K.x + (size_t)t * B_N * X_N;

    // S0: xts = repeat(x_t, P) -> f16
    for (int i = gtid; i < M_N * X_N; i += GTHREADS) {
      int r = i / X_N, c = i - r * X_N;
      AH[AH_XTS + i] = (f16)xt[(r / P_N) * X_N + c];
    }
    grid_barrier(K.bar);

    // S1: a1 = relu(xts@Wx1^T+b) ; prior_h = relu(h@priorW^T+b)
    gemm_run(lane, wave, H_N, Seg{AH+AH_XTS,  W+OFF_Wx1,    X_N, X_N}, Seg{0,0,0,0}, Seg{0,0,0,0},
             K.bx1, EPI_RELU, AH+AH_A1, nullptr, H_N);
    gemm_run(lane, wave, H_N, Seg{AH+AH_HCUR, W+OFF_priorW, H_N, H_N}, Seg{0,0,0,0}, Seg{0,0,0,0},
             K.priorB, EPI_RELU, AH+AH_PRIORH, nullptr, H_N);
    grid_barrier(K.bar);

    // S2: phi_x ; prior_mu ; prior_std
    gemm_run(lane, wave, H_N, Seg{AH+AH_A1, W+OFF_Wx2, H_N, H_N}, Seg{0,0,0,0}, Seg{0,0,0,0},
             K.bx2, EPI_RELU, AH+AH_PHIX, nullptr, H_N);
    gemm_run(lane, wave, Z_N, Seg{AH+AH_PRIORH, W+OFF_priorMean, H_N, H_N}, Seg{0,0,0,0}, Seg{0,0,0,0},
             K.priorMeanB, EPI_F32, nullptr, AF+AF_PRIMU, Z_N);
    gemm_run(lane, wave, Z_N, Seg{AH+AH_PRIORH, W+OFF_priorStd, H_N, H_N}, Seg{0,0,0,0}, Seg{0,0,0,0},
             K.priorStdB, EPI_SPLUS, nullptr, AF+AF_PRISTD, Z_N);
    grid_barrier(K.bar);

    // S3: enc_h1 = relu([phi_x,h]@encW1^T+b)  (two K segments, no concat)
    gemm_run(lane, wave, H_N,
             Seg{AH+AH_PHIX, W+OFF_encW1,       2*H_N, H_N},
             Seg{AH+AH_HCUR, W+OFF_encW1 + H_N, 2*H_N, H_N}, Seg{0,0,0,0},
             K.encB1, EPI_RELU, AH+AH_ENCH1, nullptr, H_N);
    grid_barrier(K.bar);

    // S4: enc_h
    gemm_run(lane, wave, H_N, Seg{AH+AH_ENCH1, W+OFF_encW2, H_N, H_N}, Seg{0,0,0,0}, Seg{0,0,0,0},
             K.encB2, EPI_RELU, AH+AH_ENCH, nullptr, H_N);
    grid_barrier(K.bar);

    // S5: enc_mu ; enc_std
    gemm_run(lane, wave, Z_N, Seg{AH+AH_ENCH, W+OFF_encMean, H_N, H_N}, Seg{0,0,0,0}, Seg{0,0,0,0},
             K.encMeanB, EPI_F32, nullptr, AF+AF_ENCMU, Z_N);
    gemm_run(lane, wave, Z_N, Seg{AH+AH_ENCH, W+OFF_encStd, H_N, H_N}, Seg{0,0,0,0}, Seg{0,0,0,0},
             K.encStdB, EPI_SPLUS, nullptr, AF+AF_ENCSTD, Z_N);
    grid_barrier(K.bar);

    // S6: z = mu + std*eps (f16) ; per-row KL
    for (int i = gtid; i < M_N * Z_N; i += GTHREADS) {
      float eps = nrmf(hash3(t, i, 0x1111u), hash3(t, i, 0x2222u));
      AH[AH_Z + i] = (f16)(AF[AF_ENCMU + i] + AF[AF_ENCSTD + i] * eps);
    }
    for (int r = gtid; r < M_N; r += GTHREADS) {
      float s = 0.f;
      for (int j = 0; j < Z_N; ++j) {
        float m1 = AF[AF_ENCMU + r*Z_N + j],  s1 = AF[AF_ENCSTD + r*Z_N + j];
        float m2 = AF[AF_PRIMU + r*Z_N + j],  s2 = AF[AF_PRISTD + r*Z_N + j];
        float d = m1 - m2;
        s += __logf(s2) - __logf(s1) + (s1*s1 + d*d) / (2.0f * s2 * s2) - 0.5f;
      }
      AF[AF_KL + r] = s;
    }
    grid_barrier(K.bar);

    // S7: phi_z
    gemm_run(lane, wave, H_N, Seg{AH+AH_Z, W+OFF_Wz, Z_N, Z_N}, Seg{0,0,0,0}, Seg{0,0,0,0},
             K.bz, EPI_RELU, AH+AH_PHIZ, nullptr, H_N);
    grid_barrier(K.bar);

    // S8: dec_h1 ; LSTM gates = [phi_x,phi_z]@Wih^T + h@Whh^T (3 segments)
    gemm_run(lane, wave, H_N,
             Seg{AH+AH_PHIZ, W+OFF_decW1,       2*H_N, H_N},
             Seg{AH+AH_HCUR, W+OFF_decW1 + H_N, 2*H_N, H_N}, Seg{0,0,0,0},
             K.decB1, EPI_RELU, AH+AH_DECH1, nullptr, H_N);
    gemm_run(lane, wave, 4*H_N,
             Seg{AH+AH_PHIX, W+OFF_Wih,       2*H_N, H_N},
             Seg{AH+AH_PHIZ, W+OFF_Wih + H_N, 2*H_N, H_N},
             Seg{AH+AH_HCUR, W+OFF_Whh,       H_N,   H_N},
             nullptr, EPI_F32, nullptr, AF+AF_GATES, 4*H_N);
    grid_barrier(K.bar);

    // S9: dec_h ; LSTM elementwise -> h_new(f16), c_new(f32)
    gemm_run(lane, wave, H_N, Seg{AH+AH_DECH1, W+OFF_decW2, H_N, H_N}, Seg{0,0,0,0}, Seg{0,0,0,0},
             K.decB2, EPI_RELU, AH+AH_DECH, nullptr, H_N);
    for (int i = gtid; i < M_N * H_N; i += GTHREADS) {
      int r = i / H_N, j = i - r * H_N;
      const float* g = AF + AF_GATES + (size_t)r * 4 * H_N;
      float c = sigf(g[H_N + j]) * AF[AF_CCUR + i] + sigf(g[j]) * tanhf(g[2*H_N + j]);
      AF[AF_CNEW + i] = c;
      AH[AH_HNEW + i] = (f16)(sigf(g[3*H_N + j]) * tanhf(c));
    }
    grid_barrier(K.bar);

    // S10: theta = sigmoid(dec_h @ dec_mean^T + b)
    gemm_run(lane, wave, X_N, Seg{AH+AH_DECH, W+OFF_decMean, H_N, H_N}, Seg{0,0,0,0}, Seg{0,0,0,0},
             K.decMeanB, EPI_SIG, nullptr, AF+AF_THETA, X_N);
    grid_barrier(K.bar);

    // S11: per-row Bernoulli NLL
    for (int r = gtid; r < M_N; r += GTHREADS) {
      const float* xr = xt + (r / P_N) * X_N;
      float s = 0.f;
      for (int j = 0; j < X_N; ++j) {
        float th = AF[AF_THETA + r*X_N + j];
        th = fminf(fmaxf(th, 1e-7f), 1.0f - 1e-7f);
        s -= xr[j] * __logf(th) + (1.0f - xr[j]) * __logf(1.0f - th);
      }
      AF[AF_NLL + r] = s;
    }
    grid_barrier(K.bar);

    // S12: particle bookkeeping (WG0 only)
    if (blockIdx.x == 0) {
      if (threadIdx.x == 0) {
        float s = 0.f;
        for (int r = 0; r < M_N; ++r) s += AF[AF_KL + r];
        s_klmean = s / (float)M_N;
      }
      __syncthreads();
      if (threadIdx.x < B_N) {
        int b = threadIdx.x;
        float m_t = K.mask[t * B_N + b];
        float* lwb = AF + AF_LW + b * P_N;
        float la[P_N];
        #pragma unroll
        for (int p = 0; p < P_N; ++p)
          la[p] = -(AF[AF_NLL + b*P_N + p] + AF[AF_KL + b*P_N + p]) * m_t;
        float lse_la = lse16(la, 1.0f);
        #pragma unroll
        for (int p = 0; p < P_N; ++p) lwb[p] += la[p];
        float lse1 = lse16(lwb, 1.0f);
        float lse2 = lse16(lwb, 2.0f);
        float logess = 2.0f * lse1 - lse2;
        AF[AF_IWAE + b]  += (lse_la - LOGP) * m_t;
        AF[AF_KLACC + b] += s_klmean * m_t;
        bool should = (logess <= THRESH) && (m_t > 0.0f);
        for (int p = 0; p < P_N; ++p) {
          int sel = b * P_N + p;
          if (should) {   // Gumbel-max categorical over lw[b]
            float best = -1e30f; int bi = 0;
            for (int q = 0; q < P_N; ++q) {
              float u = u01(hash3(t, b * 256 + p * P_N + q, 0xC0FFEEu));
              float sc = lwb[q] - __logf(-__logf(u));
              if (sc > best) { best = sc; bi = q; }
            }
            sel = b * P_N + bi;
          }
          K.idx[b * P_N + p] = sel;
        }
        if (should) {
          AF[AF_LHP + b] += lse1 - LOGP;
          #pragma unroll
          for (int q = 0; q < P_N; ++q) lwb[q] = 0.0f;
        }
      }
    }
    grid_barrier(K.bar);

    // S13: resample gather h,c
    for (int i = gtid; i < M_N * H_N; i += GTHREADS) {
      int r = i / H_N, j = i - r * H_N;
      int s = K.idx[r];
      AH[AH_HCUR + i] = AH[AH_HNEW + (size_t)s * H_N + j];
      AF[AF_CCUR + i] = AF[AF_CNEW + (size_t)s * H_N + j];
    }
    grid_barrier(K.bar);
  }

  // ---- finalize (WG0) ----
  if (blockIdx.x == 0) {
    if (threadIdx.x < B_N) {
      int b = threadIdx.x;
      AF[AF_LHP + b] += lse16(AF + AF_LW + b * P_N, 1.0f) - 2.7725887f;
    }
    __syncthreads();
    if (threadIdx.x == 0) {
      float s = 0.f;
      for (int b = 0; b < B_N; ++b) s += AF[AF_LHP + b];
      K.out[0] = -s;
      for (int b = 0; b < B_N; ++b) K.out[1 + b] = AF[AF_LHP + b];
      for (int i = 0; i < M_N; ++i) K.out[33 + i] = AF[AF_LW + i];
      for (int b = 0; b < B_N; ++b) K.out[545 + b] = AF[AF_KLACC + b];
      for (int b = 0; b < B_N; ++b) K.out[577 + b] = AF[AF_IWAE + b];
    }
  }
}

__global__ void cvt_f32_to_f16(const float* __restrict__ src, f16* __restrict__ dst, int n) {
  for (int i = blockIdx.x * blockDim.x + threadIdx.x; i < n; i += gridDim.x * blockDim.x)
    dst[i] = (f16)src[i];
}

extern "C" void kernel_launch(void* const* d_in, const int* in_sizes, int n_in,
                              void* d_out, int out_size, void* d_ws, size_t ws_size,
                              hipStream_t stream) {
  (void)in_sizes; (void)n_in; (void)out_size; (void)ws_size;
  // workspace carve-up
  f16* W = (f16*)d_ws;
  size_t off = (size_t)W_TOTAL * sizeof(f16);
  off = (off + 255) & ~(size_t)255;
  f16* AH = (f16*)((char*)d_ws + off);
  off += (size_t)AH_TOTAL * sizeof(f16);  off = (off + 255) & ~(size_t)255;
  float* AF = (float*)((char*)d_ws + off);
  off += (size_t)AF_TOTAL * sizeof(float); off = (off + 255) & ~(size_t)255;
  int* idx = (int*)((char*)d_ws + off);
  off += (size_t)M_N * sizeof(int);        off = (off + 255) & ~(size_t)255;
  unsigned* bar = (unsigned*)((char*)d_ws + off);

  hipMemsetAsync(bar, 0, 2 * sizeof(unsigned), stream);

  // convert weight matrices f32 -> f16 (biases stay f32, read directly)
  const int  srcIdx[15] = { 3, 5, 7, 9, 11, 13, 15, 17, 19, 21, 23, 25, 27, 29, 30 };
  const int  dstOff[15] = { OFF_Wx1, OFF_Wx2, OFF_Wz, OFF_encW1, OFF_encW2,
                            OFF_encMean, OFF_encStd, OFF_priorW, OFF_priorMean, OFF_priorStd,
                            OFF_decW1, OFF_decW2, OFF_decMean, OFF_Wih, OFF_Whh };
  const int  cnts[15]   = { H_N*X_N, H_N*H_N, H_N*Z_N, H_N*2*H_N, H_N*H_N,
                            Z_N*H_N, Z_N*H_N, H_N*H_N, Z_N*H_N, Z_N*H_N,
                            H_N*2*H_N, H_N*H_N, X_N*H_N, 4*H_N*2*H_N, 4*H_N*H_N };
  for (int i = 0; i < 15; ++i)
    cvt_f32_to_f16<<<256, 256, 0, stream>>>((const float*)d_in[srcIdx[i]], W + dstOff[i], cnts[i]);

  KArgs K;
  K.x = (const float*)d_in[0];
  K.mask = (const float*)d_in[1];
  K.bx1 = (const float*)d_in[4];   K.bx2 = (const float*)d_in[6];
  K.bz  = (const float*)d_in[8];
  K.encB1 = (const float*)d_in[10]; K.encB2 = (const float*)d_in[12];
  K.encMeanB = (const float*)d_in[14]; K.encStdB = (const float*)d_in[16];
  K.priorB = (const float*)d_in[18]; K.priorMeanB = (const float*)d_in[20];
  K.priorStdB = (const float*)d_in[22];
  K.decB1 = (const float*)d_in[24]; K.decB2 = (const float*)d_in[26];
  K.decMeanB = (const float*)d_in[28];
  K.W = W; K.AH = AH; K.AF = AF; K.idx = idx; K.bar = bar;
  K.out = (float*)d_out;

  fivo_main<<<NWG, TPB, 0, stream>>>(K);
}
